// EPMoE_66743791780447
// MI455X (gfx1250) — matrix operand order repaired and checked by reading, not compiled, not run on hardware
//
#include <hip/hip_runtime.h>
#include <hip/hip_bf16.h>
#include <math.h>

// Problem constants (match reference)
#define E_ 8
#define K_ 2
#define T_ 2048
#define H_ 2048
#define I_ 1408
#define NPAIR (T_ * K_)              // 4096 (token, k) pairs
#define PADMAX (NPAIR + E_ * 16)     // 4224 worst-case padded rows
#define MAXTILES (PADMAX / 16)       // 264 row tiles of 16

typedef __attribute__((ext_vector_type(16))) __bf16 v16bf;
typedef __attribute__((ext_vector_type(4)))  __bf16 v4bf;
typedef __attribute__((ext_vector_type(8)))  float  v8f;

#define LDS_STRIDE 136               // 128 K-chunk + 8 pad (rows stay 16B-aligned: 272B)

// ---------------- workspace layout (bytes) ----------------
static constexpr size_t OFF_OFF  = 0;
static constexpr size_t OFF_SORT = 64;
static constexpr size_t OFF_SLOT = OFF_SORT + (size_t)PADMAX * 4;
static constexpr size_t OFF_H    = ((OFF_SLOT + (size_t)NPAIR * 4 + 255) / 256) * 256;
static constexpr size_t OFF_O2   = OFF_H + (size_t)PADMAX * I_ * 4;

__device__ __forceinline__ __bf16 f2bf(float f) { return static_cast<__bf16>(f); }

// 4 floats -> 4 bf16 (lowers to paired v_cvt_pk_bf16_f32 + ds_store_b64)
__device__ __forceinline__ v4bf cvt4(float4 v) {
  v4bf r;
  r[0] = f2bf(v.x); r[1] = f2bf(v.y); r[2] = f2bf(v.z); r[3] = f2bf(v.w);
  return r;
}

// ---------------- routing: counting sort by expert, 16-padded segments ----------------
__global__ __launch_bounds__(256) void moe_route(const int* __restrict__ topk_ids,
                                                 int* __restrict__ off,
                                                 int* __restrict__ sorted_row,
                                                 int* __restrict__ slot_of) {
  __shared__ int cnt[E_];
  __shared__ int cur[E_];
  __shared__ int off_s[E_ + 1];
  const int tid = threadIdx.x;
  if (tid < E_) { cnt[tid] = 0; cur[tid] = 0; }
  for (int p = tid; p < PADMAX; p += 256) sorted_row[p] = -1;  // padding marker
  __syncthreads();
  for (int i = tid; i < NPAIR; i += 256) atomicAdd(&cnt[topk_ids[i]], 1);
  __syncthreads();
  if (tid == 0) {
    int run = 0;
    for (int e = 0; e < E_; ++e) { off_s[e] = run; run += (cnt[e] + 15) & ~15; }
    off_s[E_] = run;
  }
  __syncthreads();
  if (tid <= E_) off[tid] = off_s[tid];
  for (int i = tid; i < NPAIR; i += 256) {
    int e = topk_ids[i];
    int p = off_s[e] + atomicAdd(&cur[e], 1);
    sorted_row[p] = i;        // i = t*K + k
    slot_of[i] = p;
  }
}

// ---------------- bf16 fragment builders (ISA 7.12.2 16-bit layouts) ----------------
// A 16x32: lane m = lane&15; comps 0-7 -> K = kb..kb+7, comps 8-15 -> K = kb+16..kb+23,
//          kb = (lane<16) ? 0 : 8.
__device__ __forceinline__ v16bf load_a_frag(const __bf16* row, int kk, int lane) {
  const int kb = (lane < 16) ? 0 : 8;
  v16bf a;
#pragma unroll
  for (int c = 0; c < 8; ++c) {
    a[c]     = row[kk + kb + c];
    a[c + 8] = row[kk + kb + 16 + c];
  }
  return a;
}
// B 32x16: lane n = lane&15; comps 0-7 -> K = kbB..kbB+7, comps 8-15 -> K = kbB+16..kbB+23,
//          kbB = (lane<16) ? 0 : 8.
__device__ __forceinline__ v16bf load_b_frag(const __bf16* row, int kk, int lane) {
  const int kbB = (lane < 16) ? 0 : 8;
  v16bf b;
#pragma unroll
  for (int c = 0; c < 8; ++c) {
    b[c]     = row[kk + kbB + c];
    b[c + 8] = row[kk + kbB + 16 + c];
  }
  return b;
}

// ---------------- GEMM1: 16 tokens x 64 intermediate cols; gate+up + SiLU*mul ----------------
// 8 waves: waves 0-3 compute gate strips, waves 4-7 the matching up strips.
__global__ __launch_bounds__(256) void moe_gemm1(const float* __restrict__ x,
                                                 const float* __restrict__ topk_w,
                                                 const float* __restrict__ w13,
                                                 const int* __restrict__ off,
                                                 const int* __restrict__ sorted_row,
                                                 float* __restrict__ h_ws) {
  __shared__ __bf16 As[16][LDS_STRIDE];
  __shared__ __bf16 Bg[64][LDS_STRIDE];
  __shared__ __bf16 Bu[64][LDS_STRIDE];
  __shared__ int    s_tok[16];
  __shared__ float  s_scale[16];
  __shared__ int    s_off[E_ + 1];

  const int tid = threadIdx.x;
  if (tid <= E_) s_off[tid] = off[tid];
  __syncthreads();

  const int row0 = blockIdx.x * 16;
  if (row0 >= s_off[E_]) return;
  int e = 0;
  while (row0 >= s_off[e + 1]) ++e;

  if (tid < 16) {
    int i = sorted_row[row0 + tid];
    s_tok[tid]   = (i >= 0) ? (i >> 1) : 0;        // token id (K=2)
    s_scale[tid] = (i >= 0) ? topk_w[i] : 0.0f;    // fold routing weight into h
  }
  __syncthreads();

  const int i0 = blockIdx.y * 64;  // intermediate-column base
  const float* __restrict__ wg = w13 + (size_t)e * (2 * I_) * H_ + (size_t)i0 * H_;
  const float* __restrict__ wu = wg + (size_t)I_ * H_;

  const int wid = tid >> 5, lane = tid & 31;
  const int m  = lane & 15;
  const int nl = (wid & 3) * 16 + (lane & 15);

  v8f acc = {};
  for (int k0 = 0; k0 < H_; k0 += 128) {      // 16 K-chunks over H
    __syncthreads();
    // stage A: 16 rows x 128 = 512 float4, 2 per thread (gathered rows, coalesced cols)
#pragma unroll
    for (int j = 0; j < 2; ++j) {
      int idx = tid + j * 256;
      int r = idx >> 5, q = idx & 31;
      float4 v = *(const float4*)&x[(size_t)s_tok[r] * H_ + k0 + q * 4];
      *(v4bf*)&As[r][q * 4] = cvt4(v);
    }
    // stage B: gate + up, 64 rows x 128 each = 2048 float4 each, 8 per thread
#pragma unroll
    for (int j = 0; j < 8; ++j) {
      int idx = tid + j * 256;
      int n = idx >> 5, q = idx & 31;
      *(v4bf*)&Bg[n][q * 4] = cvt4(*(const float4*)&wg[(size_t)n * H_ + k0 + q * 4]);
      *(v4bf*)&Bu[n][q * 4] = cvt4(*(const float4*)&wu[(size_t)n * H_ + k0 + q * 4]);
    }
    if (k0 + 128 < H_ && tid < 64) {  // hint next weight chunk
      __builtin_prefetch(&wg[(size_t)tid * H_ + k0 + 128], 0, 1);
      __builtin_prefetch(&wu[(size_t)tid * H_ + k0 + 128], 0, 1);
    }
    __syncthreads();

    const __bf16 (*B)[LDS_STRIDE] = (wid < 4) ? Bg : Bu;
#pragma unroll
    for (int kk = 0; kk < 128; kk += 32) {
      v16bf a = load_a_frag(As[m], kk, lane);
      v16bf b = load_b_frag(B[nl], kk, lane);
      acc = __builtin_amdgcn_wmma_f32_16x16x32_bf16(false, a, false, b, (short)0, acc,
                                                    false, false);
    }
  }
  __syncthreads();

  // waves 4-7 publish their "up" accumulators via LDS (reuse Bu space as float)
  float* U = (float*)&Bu[0][0];  // 4 * 16 * 16 floats = 4 KB (fits in Bu)
  if (wid >= 4) {
    const int w = wid - 4;
#pragma unroll
    for (int r = 0; r < 8; ++r) {
      int mm = (lane < 16) ? r : (r + 8);
      U[(w * 16 + mm) * 16 + (lane & 15)] = acc[r];
    }
  }
  __syncthreads();
  if (wid < 4) {
#pragma unroll
    for (int r = 0; r < 8; ++r) {
      int mm = (lane < 16) ? r : (r + 8);
      int nn = lane & 15;
      float g = acc[r];
      float u = U[(wid * 16 + mm) * 16 + nn];
      float hv = (g / (1.0f + __expf(-g))) * u * s_scale[mm];  // silu(g)*u, weight folded
      h_ws[(size_t)(row0 + mm) * I_ + i0 + wid * 16 + nn] = hv;
    }
  }
}

// ---------------- GEMM2: 16 rows x 128 hidden cols over K=I ----------------
__global__ __launch_bounds__(256) void moe_gemm2(const float* __restrict__ h_ws,
                                                 const float* __restrict__ w2,
                                                 const int* __restrict__ off,
                                                 float* __restrict__ out2) {
  __shared__ __bf16 As[16][LDS_STRIDE];
  __shared__ __bf16 Bs[128][LDS_STRIDE];
  __shared__ int    s_off[E_ + 1];

  const int tid = threadIdx.x;
  if (tid <= E_) s_off[tid] = off[tid];
  __syncthreads();

  const int row0 = blockIdx.x * 16;
  if (row0 >= s_off[E_]) return;
  int e = 0;
  while (row0 >= s_off[e + 1]) ++e;

  const int n0 = blockIdx.y * 128;
  const float* __restrict__ w2e = w2 + (size_t)e * H_ * I_ + (size_t)n0 * I_;

  const int wid = tid >> 5, lane = tid & 31;
  const int m  = lane & 15;
  const int nl = wid * 16 + (lane & 15);

  v8f acc = {};
  for (int k0 = 0; k0 < I_; k0 += 128) {      // 11 K-chunks over I
    __syncthreads();
    // stage A: 16 rows x 128 = 512 float4, 2 per thread
#pragma unroll
    for (int j = 0; j < 2; ++j) {
      int idx = tid + j * 256;
      int r = idx >> 5, q = idx & 31;
      float4 v = *(const float4*)&h_ws[(size_t)(row0 + r) * I_ + k0 + q * 4];
      *(v4bf*)&As[r][q * 4] = cvt4(v);
    }
    // stage B: 128 rows x 128 = 4096 float4, 16 per thread
#pragma unroll
    for (int j = 0; j < 16; ++j) {
      int idx = tid + j * 256;
      int n = idx >> 5, q = idx & 31;
      *(v4bf*)&Bs[n][q * 4] = cvt4(*(const float4*)&w2e[(size_t)n * I_ + k0 + q * 4]);
    }
    if (k0 + 128 < I_ && tid < 128)
      __builtin_prefetch(&w2e[(size_t)tid * I_ + k0 + 128], 0, 1);
    __syncthreads();

#pragma unroll
    for (int kk = 0; kk < 128; kk += 32) {
      v16bf a = load_a_frag(As[m], kk, lane);
      v16bf b = load_b_frag(Bs[nl], kk, lane);
      acc = __builtin_amdgcn_wmma_f32_16x16x32_bf16(false, a, false, b, (short)0, acc,
                                                    false, false);
    }
  }

#pragma unroll
  for (int r = 0; r < 8; ++r) {
    int mm = (lane < 16) ? r : (r + 8);
    int nn = lane & 15;
    out2[(size_t)(row0 + mm) * H_ + n0 + wid * 16 + nn] = acc[r];
  }
}

// ---------------- combine: out[t] = out2[slot(t,0)] + out2[slot(t,1)], float4 ----------------
__global__ __launch_bounds__(256) void moe_combine(const float* __restrict__ out2,
                                                   const int* __restrict__ slot_of,
                                                   float* __restrict__ out) {
  constexpr int QTOT = T_ * H_ / 4;   // float4 elements
  constexpr int QH   = H_ / 4;        // 512 float4 per token
  int idx = blockIdx.x * 256 + threadIdx.x;
  const int stride = gridDim.x * 256;
  for (; idx < QTOT; idx += stride) {
    int t = idx / QH;
    int c = idx - t * QH;
    const float4* r0 = (const float4*)(out2 + (size_t)slot_of[t * K_ + 0] * H_);
    const float4* r1 = (const float4*)(out2 + (size_t)slot_of[t * K_ + 1] * H_);
    float4 a = r0[c], b = r1[c];
    float4 s; s.x = a.x + b.x; s.y = a.y + b.y; s.z = a.z + b.z; s.w = a.w + b.w;
    ((float4*)out)[idx] = s;
  }
}

// ---------------- launcher ----------------
extern "C" void kernel_launch(void* const* d_in, const int* in_sizes, int n_in,
                              void* d_out, int out_size, void* d_ws, size_t ws_size,
                              hipStream_t stream) {
  (void)in_sizes; (void)n_in; (void)out_size; (void)ws_size;
  const float* x    = (const float*)d_in[0];  // hidden_states [T,H]
  const float* tw   = (const float*)d_in[1];  // topk_weights  [T,K]
  const float* w13  = (const float*)d_in[2];  // [E, 2I, H]
  const float* w2   = (const float*)d_in[3];  // [E, H, I]
  const int*   ids  = (const int*)d_in[4];    // topk_ids [T,K]
  float* out = (float*)d_out;

  char* ws = (char*)d_ws;
  int*   off    = (int*)(ws + OFF_OFF);
  int*   sorted = (int*)(ws + OFF_SORT);
  int*   slot   = (int*)(ws + OFF_SLOT);
  float* h      = (float*)(ws + OFF_H);
  float* o2     = (float*)(ws + OFF_O2);

  moe_route<<<1, 256, 0, stream>>>(ids, off, sorted, slot);
  moe_gemm1<<<dim3(MAXTILES, I_ / 64), 256, 0, stream>>>(x, tw, w13, off, sorted, h);
  moe_gemm2<<<dim3(MAXTILES, H_ / 128), 256, 0, stream>>>(h, w2, off, o2);
  moe_combine<<<dim3(2048), 256, 0, stream>>>(o2, slot, out);
}
